// GraphDynamics_69853348102720
// MI455X (gfx1250) — compile-verified, matrix operand order checked by reading, not run on hardware
//
#include <hip/hip_runtime.h>

// ---------------------------------------------------------------------------
// GraphDynamics fused message-passing for gfx1250 (MI455X), wave32 + WMMA.
//   B=32, N=128, D=128, H=256.  Dominant GEMM: [B*N*N,H]x[H,D] ~34 GFLOP,
//   fully fused (the [B,N,N,H] tensor never exists). f16 inputs, f32 acc via
//   v_wmma_f32_16x16x32_f16.  Hot weight operand staged into LDS with the
//   Tensor Data Mover (tensor_load_to_lds + s_wait_tensorcnt).
// ---------------------------------------------------------------------------

typedef __attribute__((ext_vector_type(16))) _Float16     v16h;
typedef __attribute__((ext_vector_type(8)))  float        v8f;
typedef __attribute__((ext_vector_type(4)))  unsigned int u32x4;
typedef __attribute__((ext_vector_type(8)))  int          i32x8;
typedef __attribute__((ext_vector_type(4)))  int          i32x4;

constexpr int Bc = 32, Nc = 128, Dc = 128, Hc = 256, Rc = Bc * Nc;  // Rc=4096

// --- WMMA fragment index helpers (16-bit A 16x32 layout, ISA 7.12.2) -------
// lane L = m + 16*hk ; element e: k = 8*hk+e (e<8) else 16+8*hk+(e-8)
__device__ __forceinline__ int kmap(int hk, int e) {
  return (e < 8) ? (hk * 8 + e) : (16 + hk * 8 + (e - 8));
}
// inverse: where does logical k land inside a packed row of 32-element (2-halves) k-step blocks
__device__ __forceinline__ int kpackoff(int k) {
  int s = k >> 5, kk = k & 31, hk, e;
  if (kk < 16) { hk = kk >> 3; e = kk & 7; }
  else         { int t = kk - 16; hk = t >> 3; e = 8 + (t & 7); }
  return s * 32 + hk * 16 + e;
}

__device__ __forceinline__ v8f wmma32(v16h a, v16h b, v8f c) {
  return __builtin_amdgcn_wmma_f32_16x16x32_f16(
      /*neg_a=*/false, a, /*neg_b=*/false, b,
      /*c_mod=*/(short)0, c, /*reuse_a=*/false, /*reuse_b=*/false);
}

// ---------------------------------------------------------------------------
// Kernel 1: LayerNorm(nodes)*mask -> packed f16 rows (fragment-ready layout)
// ---------------------------------------------------------------------------
__global__ void k_ln_pack(const float* __restrict__ nodes,
                          const float* __restrict__ mask,
                          const float* __restrict__ g,
                          const float* __restrict__ bvec,
                          _Float16* __restrict__ xpk) {
  int lane = threadIdx.x & 31, wave = threadIdx.x >> 5;
  int row  = blockIdx.x * 8 + wave;
  const float4 v4 = ((const float4*)(nodes + (size_t)row * Dc))[lane];
  float vv[4] = {v4.x, v4.y, v4.z, v4.w};
  float s = vv[0] + vv[1] + vv[2] + vv[3];
#pragma unroll
  for (int o = 16; o > 0; o >>= 1) s += __shfl_xor(s, o);
  float mu = s * (1.0f / Dc);
  float q = 0.f;
#pragma unroll
  for (int t = 0; t < 4; t++) { float d = vv[t] - mu; q += d * d; }
#pragma unroll
  for (int o = 16; o > 0; o >>= 1) q += __shfl_xor(q, o);
  float rs = rsqrtf(q * (1.0f / Dc) + 1e-5f);
  float m  = mask[row];
  int d0 = lane * 4, po = kpackoff(d0);   // 4 consecutive d stay contiguous
#pragma unroll
  for (int t = 0; t < 4; t++) {
    int d = d0 + t;
    float y = (vv[t] - mu) * rs * g[d] + bvec[d];
    xpk[(size_t)row * Dc + po + t] = (_Float16)(y * m);
  }
}

// ---------------------------------------------------------------------------
// Kernel 2: pack a [K,Nw] f32 weight into B-fragment layout:
//   bpk[ntile][kstep][lane][e] = W[kstep*32 + kmap(lane>>4,e)][ntile*16+lane&15]
// ---------------------------------------------------------------------------
__global__ void k_pack_b(const float* __restrict__ W, _Float16* __restrict__ bpk,
                         int K, int Nw) {
  int idx = blockIdx.x * blockDim.x + threadIdx.x;
  if (idx >= K * Nw) return;
  int e = idx & 15, lane = (idx >> 4) & 31, rest = idx >> 9;
  int ks = K >> 5;
  int kstep = rest % ks, ntile = rest / ks;
  int n = ntile * 16 + (lane & 15);
  int k = kstep * 32 + kmap(lane >> 4, e);
  bpk[idx] = (_Float16)W[(size_t)k * Nw + n];
}

// pack f32 rows [R,C] -> packed f16 rows (A-fragment ready); also for biases
__global__ void k_pack_rows(const float* __restrict__ src, _Float16* __restrict__ dst,
                            int C, int total) {
  int idx = blockIdx.x * blockDim.x + threadIdx.x;
  if (idx >= total) return;
  int row = idx / C, c = idx % C;
  dst[(size_t)row * C + kpackoff(c)] = (_Float16)src[idx];
}

// ---------------------------------------------------------------------------
// Kernel 3: projections xi_p = x@mw1[:D], xj_p = x@mw1[D:]  (WMMA, K=128)
// ---------------------------------------------------------------------------
__global__ void k_proj(const _Float16* __restrict__ xpk,
                       const _Float16* __restrict__ mw1apk,
                       const _Float16* __restrict__ mw1bpk,
                       _Float16* __restrict__ xipk,
                       _Float16* __restrict__ xjpk) {
  int lane = threadIdx.x & 31, wave = threadIdx.x >> 5;
  int hk = lane >> 4, nlo = lane & 15;
  int job = blockIdx.x * 8 + wave;          // 8192 jobs
  int which = job >> 12, rem = job & 4095;
  int mtile = rem >> 4, ntile = rem & 15;
  const _Float16* bpk  = which ? mw1bpk : mw1apk;
  const _Float16* arow = xpk + (size_t)(mtile * 16 + nlo) * Dc;
  v8f c = {};
#pragma unroll
  for (int s = 0; s < 4; s++) {
    v16h a = *(const v16h*)(arow + s * 32 + hk * 16);
    v16h b = *(const v16h*)(bpk + ((size_t)(ntile * 4 + s) * 32 + lane) * 16);
    c = wmma32(a, b, c);
  }
  _Float16* outp = which ? xjpk : xipk;
  int nn = ntile * 16 + nlo, po = kpackoff(nn);
#pragma unroll
  for (int v = 0; v < 8; v++)
    outp[(size_t)(mtile * 16 + v + 8 * hk) * Hc + po] = (_Float16)c[v];
}

// ---------------------------------------------------------------------------
// Kernel 4 (MAIN): fused edge MLP + masked sender-sum.
// One wave per receiver i. mw2 (64KB, shared by all 8 waves) is DMA'd into
// LDS once per workgroup by the Tensor Data Mover; inner loop reads B
// fragments via ds_load, keeping the VMEM pipe free for streaming xj rows.
// Per j-tile: A = relu(xi+xj+mb1) built in registers (v_pk f16 ops), then
// 8 d-tiles x 8 k-steps of WMMA; relu+bias+mask epilogue folds the sum over
// senders into a tile row-reduction + shfl_xor(16). 512 WMMAs per wave.
// ---------------------------------------------------------------------------
__global__ void k_msg(const _Float16* __restrict__ xipk,
                      const _Float16* __restrict__ xjpk,
                      const _Float16* __restrict__ mb1pk,
                      const _Float16* __restrict__ mw2pk,
                      const float* __restrict__ mb2,
                      const float* __restrict__ mask,
                      float* __restrict__ agg) {
  __shared__ _Float16 lds_w2[Hc * Dc];      // 64 KB packed mw2 fragments

  int lane = threadIdx.x & 31, wave = threadIdx.x >> 5;
  int hk = lane >> 4, nlo = lane & 15;

  // --- TDM: one DMA of the whole packed weight block into LDS (wave 0) ----
  if (wave == 0) {
    unsigned ldsoff = (unsigned)(uintptr_t)&lds_w2[0];      // low 32 bits = LDS offset
    unsigned long long ga = (unsigned long long)(uintptr_t)mw2pk;
    u32x4 g0;
    g0[0] = 1u;                                             // count=1 valid descriptor
    g0[1] = ldsoff;                                         // lds_addr
    g0[2] = (unsigned)(ga & 0xffffffffu);                   // global_addr[31:0]
    g0[3] = (unsigned)((ga >> 32) & 0x1ffffffu) | (2u << 30); // addr[56:32] | type=2
    i32x8 g1;
    g1[0] = (3 << 16);                 // data_size = 8 bytes
    g1[1] = (int)(8192u << 16);        // tensor_dim0 = 8192 elems (low 16 in [31:16])
    g1[2] = (int)(1u << 16);           // tensor_dim0 hi=0 ; tensor_dim1 = 1
    g1[3] = (int)(8192u << 16);        // tensor_dim1 hi=0 ; tile_dim0 = 8192
    g1[4] = 1;                         // tile_dim1 = 1 ; tile_dim2 = 0
    g1[5] = 8192;                      // tensor_dim0_stride low32
    g1[6] = (int)(8192u << 16);        // stride hi=0 ; tensor_dim1_stride low16
    g1[7] = 0;
    i32x4 g2 = {0, 0, 0, 0};           // <=2D tensor: groups 2/3 unused
    i32x4 g3 = {0, 0, 0, 0};
    i32x8 g4 = {0, 0, 0, 0, 0, 0, 0, 0};
    __builtin_amdgcn_tensor_load_to_lds(g0, g1, g2, g3, g4, 0);
    __builtin_amdgcn_s_wait_tensorcnt(0);
  }
  __syncthreads();

  int gi = blockIdx.x * 8 + wave;           // global receiver row 0..4095
  int b  = gi >> 7;
  const _Float16* xirow  = xipk + (size_t)gi * Hc;
  const _Float16* xjbase = xjpk + (size_t)b * Nc * Hc;
  const float*    maskb  = mask + b * Nc;

  // hoist xi + mb1 (receiver part of h), fragment layout, 8 ksteps
  v16h xim[8];
#pragma unroll
  for (int s = 0; s < 8; s++) {
    v16h xi = *(const v16h*)(xirow + s * 32 + hk * 16);
    v16h mb = *(const v16h*)(mb1pk + s * 32 + hk * 16);
    xim[s] = xi + mb;
  }
  float mb2v[8];
#pragma unroll
  for (int dt = 0; dt < 8; dt++) mb2v[dt] = mb2[dt * 16 + nlo];

  float aggacc[8] = {0.f, 0.f, 0.f, 0.f, 0.f, 0.f, 0.f, 0.f};
  const v16h hzero = {};

  for (int jt = 0; jt < 8; jt++) {
    // sender masks for the 8 output-tile rows this lane-half owns
    float mrow[8];
#pragma unroll
    for (int v = 0; v < 8; v++) mrow[v] = maskb[jt * 16 + v + 8 * hk];
    // A fragments: h = relu(xi + xj + mb1)  (v_pk_add_f16 / v_pk_max)
    v16h af[8];
#pragma unroll
    for (int s = 0; s < 8; s++) {
      v16h xj = *(const v16h*)(xjbase + (size_t)(jt * 16 + nlo) * Hc + s * 32 + hk * 16);
      af[s] = __builtin_elementwise_max(xim[s] + xj, hzero);
    }
#pragma unroll
    for (int dt = 0; dt < 8; dt++) {
      v8f c = {};
#pragma unroll
      for (int s = 0; s < 8; s++) {
        v16h bfr = *(const v16h*)(&lds_w2[((dt * 8 + s) * 32 + lane) * 16]);
        c = wmma32(af[s], bfr, c);
      }
      // relu(+bias) * mask[j], reduce over the 16 senders of this tile
      float p = 0.f;
#pragma unroll
      for (int v = 0; v < 8; v++) {
        float x = c[v] + mb2v[dt];
        x = (x > 0.f) ? x : 0.f;
        p += x * mrow[v];
      }
      p += __shfl_xor(p, 16);   // combine M=0..7 half with M=8..15 half
      aggacc[dt] += p;
    }
  }
  if (hk == 0) {
#pragma unroll
    for (int dt = 0; dt < 8; dt++) agg[(size_t)gi * Dc + dt * 16 + nlo] = aggacc[dt];
  }
}

// ---------------------------------------------------------------------------
// Kernel 5: u = relu(nodes@uw1[:D] + agg@uw1[D:] + ub1)  -> packed f16 rows
// ---------------------------------------------------------------------------
__global__ void k_upd1(const _Float16* __restrict__ nodespk,
                       const _Float16* __restrict__ aggpk,
                       const _Float16* __restrict__ uw1apk,
                       const _Float16* __restrict__ uw1bpk,
                       const float* __restrict__ ub1,
                       _Float16* __restrict__ upk) {
  int lane = threadIdx.x & 31, wave = threadIdx.x >> 5;
  int hk = lane >> 4, nlo = lane & 15;
  int job = blockIdx.x * 8 + wave;          // 4096 jobs
  int mtile = job >> 4, ntile = job & 15;
  const _Float16* a1 = nodespk + (size_t)(mtile * 16 + nlo) * Dc;
  const _Float16* a2 = aggpk   + (size_t)(mtile * 16 + nlo) * Dc;
  v8f c = {};
#pragma unroll
  for (int s = 0; s < 4; s++) {
    v16h a = *(const v16h*)(a1 + s * 32 + hk * 16);
    v16h b = *(const v16h*)(uw1apk + ((size_t)(ntile * 4 + s) * 32 + lane) * 16);
    c = wmma32(a, b, c);
  }
#pragma unroll
  for (int s = 0; s < 4; s++) {
    v16h a = *(const v16h*)(a2 + s * 32 + hk * 16);
    v16h b = *(const v16h*)(uw1bpk + ((size_t)(ntile * 4 + s) * 32 + lane) * 16);
    c = wmma32(a, b, c);
  }
  int nn = ntile * 16 + nlo, po = kpackoff(nn);
  float ubv = ub1[nn];
#pragma unroll
  for (int v = 0; v < 8; v++) {
    float x = c[v] + ubv;
    x = (x > 0.f) ? x : 0.f;
    upk[(size_t)(mtile * 16 + v + 8 * hk) * Hc + po] = (_Float16)x;
  }
}

// ---------------------------------------------------------------------------
// Kernel 6: out = (nodes + u@uw2 + ub2) * mask     (f32 output)
// ---------------------------------------------------------------------------
__global__ void k_upd2(const _Float16* __restrict__ upk,
                       const _Float16* __restrict__ uw2pk,
                       const float* __restrict__ ub2,
                       const float* __restrict__ nodes,
                       const float* __restrict__ mask,
                       float* __restrict__ out) {
  int lane = threadIdx.x & 31, wave = threadIdx.x >> 5;
  int hk = lane >> 4, nlo = lane & 15;
  int job = blockIdx.x * 8 + wave;          // 2048 jobs
  int mtile = job >> 3, dtile = job & 7;
  const _Float16* arow = upk + (size_t)(mtile * 16 + nlo) * Hc;
  v8f c = {};
#pragma unroll
  for (int s = 0; s < 8; s++) {
    v16h a = *(const v16h*)(arow + s * 32 + hk * 16);
    v16h b = *(const v16h*)(uw2pk + ((size_t)(dtile * 8 + s) * 32 + lane) * 16);
    c = wmma32(a, b, c);
  }
  int nn = dtile * 16 + nlo;
  float ubv = ub2[nn];
#pragma unroll
  for (int v = 0; v < 8; v++) {
    int r2 = mtile * 16 + v + 8 * hk;
    float val = nodes[(size_t)r2 * Dc + nn] + c[v] + ubv;
    out[(size_t)r2 * Dc + nn] = val * mask[r2];
  }
}

// ---------------------------------------------------------------------------
extern "C" void kernel_launch(void* const* d_in, const int* in_sizes, int n_in,
                              void* d_out, int out_size, void* d_ws, size_t ws_size,
                              hipStream_t stream) {
  (void)in_sizes; (void)n_in; (void)out_size; (void)ws_size;
  const float* nodes = (const float*)d_in[0];
  const float* nmask = (const float*)d_in[1];
  const float* ln_g  = (const float*)d_in[2];
  const float* ln_b  = (const float*)d_in[3];
  const float* mw1   = (const float*)d_in[4];
  const float* mb1   = (const float*)d_in[5];
  const float* mw2   = (const float*)d_in[6];
  const float* mb2   = (const float*)d_in[7];
  const float* uw1   = (const float*)d_in[8];
  const float* ub1   = (const float*)d_in[9];
  const float* uw2   = (const float*)d_in[10];
  const float* ub2   = (const float*)d_in[11];
  float* outp        = (float*)d_out;

  // workspace carve-up (256B aligned)
  char* w = (char*)d_ws;
  size_t off = 0;
  auto alloc = [&](size_t bytes) -> void* {
    void* p = w + off;
    off += (bytes + 255) & ~(size_t)255;
    return p;
  };
  _Float16* xpk     = (_Float16*)alloc((size_t)Rc * Dc * 2);   // 1 MB
  _Float16* xipk    = (_Float16*)alloc((size_t)Rc * Hc * 2);   // 2 MB
  _Float16* xjpk    = (_Float16*)alloc((size_t)Rc * Hc * 2);   // 2 MB
  _Float16* mw1apk  = (_Float16*)alloc((size_t)Dc * Hc * 2);
  _Float16* mw1bpk  = (_Float16*)alloc((size_t)Dc * Hc * 2);
  _Float16* mw2pk   = (_Float16*)alloc((size_t)Hc * Dc * 2);
  _Float16* uw1apk  = (_Float16*)alloc((size_t)Dc * Hc * 2);
  _Float16* uw1bpk  = (_Float16*)alloc((size_t)Dc * Hc * 2);
  _Float16* uw2pk   = (_Float16*)alloc((size_t)Hc * Dc * 2);
  _Float16* mb1pk   = (_Float16*)alloc((size_t)Hc * 2);
  float*    agg     = (float*)   alloc((size_t)Rc * Dc * 4);   // 2 MB
  _Float16* nodespk = (_Float16*)alloc((size_t)Rc * Dc * 2);
  _Float16* aggpk   = (_Float16*)alloc((size_t)Rc * Dc * 2);
  _Float16* upk     = (_Float16*)alloc((size_t)Rc * Hc * 2);

  // 1) LayerNorm + mask -> packed x
  k_ln_pack<<<Rc / 8, 256, 0, stream>>>(nodes, nmask, ln_g, ln_b, xpk);
  // 2) weight packing (fragment layouts)
  k_pack_b<<<128, 256, 0, stream>>>(mw1,            mw1apk, 128, 256);
  k_pack_b<<<128, 256, 0, stream>>>(mw1 + 128*256,  mw1bpk, 128, 256);
  k_pack_b<<<128, 256, 0, stream>>>(mw2,            mw2pk,  256, 128);
  k_pack_b<<<128, 256, 0, stream>>>(uw1,            uw1apk, 128, 256);
  k_pack_b<<<128, 256, 0, stream>>>(uw1 + 128*256,  uw1bpk, 128, 256);
  k_pack_b<<<128, 256, 0, stream>>>(uw2,            uw2pk,  256, 128);
  k_pack_rows<<<1, 256, 0, stream>>>(mb1, mb1pk, 256, 256);
  // 3) projections (WMMA)
  k_proj<<<1024, 256, 0, stream>>>(xpk, mw1apk, mw1bpk, xipk, xjpk);
  // 4) fused edge MLP + masked aggregation (the 34 GFLOP kernel, TDM-staged B)
  k_msg<<<Rc / 8, 256, 0, stream>>>(xipk, xjpk, mb1pk, mw2pk, mb2, nmask, agg);
  // 5) update MLP (WMMA) + residual + mask
  k_pack_rows<<<(Rc * Dc + 255) / 256, 256, 0, stream>>>(nodes, nodespk, 128, Rc * Dc);
  k_pack_rows<<<(Rc * Dc + 255) / 256, 256, 0, stream>>>(agg,   aggpk,   128, Rc * Dc);
  k_upd1<<<512, 256, 0, stream>>>(nodespk, aggpk, uw1apk, uw1bpk, ub1, upk);
  k_upd2<<<256, 256, 0, stream>>>(upk, uw2pk, ub2, nodes, nmask, outp);
}
